// MultiHeadAttention_11931419148341
// MI455X (gfx1250) — matrix operand order repaired and checked by reading, not compiled, not run on hardware
//
#include <hip/hip_runtime.h>

// ---------------------------------------------------------------------------
// MHA forward for MI455X (gfx1250): bf16 WMMA everywhere, f32 softmax.
// Async global->LDS double buffering in the GEMMs (ASYNCcnt path).
// ---------------------------------------------------------------------------

typedef __attribute__((ext_vector_type(16))) __bf16 v16bf;
typedef __attribute__((ext_vector_type(8)))  float  v8f;
typedef __attribute__((ext_vector_type(4)))  int    v4i;
typedef unsigned short ushort_t;

#define B_  4
#define S_  2048
#define D_  1024
#define H_  16
#define HD_ 64

#define AS1 __attribute__((address_space(1)))
#define AS3 __attribute__((address_space(3)))

#if __has_builtin(__builtin_amdgcn_global_load_async_to_lds_b128)
#define ASYNC_LDS 1
#else
#define ASYNC_LDS 0
#endif

#if ASYNC_LDS
# if __has_builtin(__builtin_amdgcn_s_wait_asynccnt)
#  define WAIT_ASYNC(N) __builtin_amdgcn_s_wait_asynccnt(N)
# else
#  define WAIT_ASYNC(N) asm volatile("s_wait_asynccnt %0" :: "n"(N) : "memory")
# endif
#else
# define WAIT_ASYNC(N) do {} while (0)
#endif

union Frag { v16bf v; uint4 q[2]; };
union Acc  { v8f v; float f[8]; };

__device__ __forceinline__ unsigned short f2bf(float x) {
    unsigned int u = __builtin_bit_cast(unsigned int, x);
    u += 0x7fffu + ((u >> 16) & 1u);           // round-to-nearest-even
    return (unsigned short)(u >> 16);
}

__device__ __forceinline__ AS3 v4i* to_lds(void* p) {
    return (AS3 v4i*)(unsigned int)(unsigned long long)p;
}
__device__ __forceinline__ AS1 v4i* to_glb(const void* p) {
    return (AS1 v4i*)(unsigned long long)p;
}

// 16-byte global -> LDS copy: async (ASYNCcnt-tracked) when available.
__device__ __forceinline__ void cp16B(ushort_t* dst_lds, const ushort_t* src_g) {
#if ASYNC_LDS
    __builtin_amdgcn_global_load_async_to_lds_b128(to_glb(src_g), to_lds(dst_lds), 0, 0);
#else
    *(uint4*)dst_lds = *(const uint4*)src_g;
#endif
}

// A-matrix fragment (16 rows x 32 K, bf16). Per ISA: lane = row (mod 16),
// lanes 0-15 hold K {0..7,16..23}, lanes 16-31 hold K {8..15,24..31}.
__device__ __forceinline__ v16bf load_fragA(const ushort_t* base, int stride, int lane) {
    const ushort_t* p = base + (size_t)(lane & 15) * stride + ((lane >> 4) << 3);
    Frag f;
    f.q[0] = *(const uint4*)(p);
    f.q[1] = *(const uint4*)(p + 16);
    return f.v;
}

// B-matrix fragment (32 K x 16 N, bf16) loaded from [N, K]-contiguous memory.
// Per ISA: lane = N (mod 16); lanes 0-15 hold K {0..7,16..23}, lanes 16-31 hold K {8..15,24..31}.
__device__ __forceinline__ v16bf load_fragB(const ushort_t* base, size_t stride, int lane) {
    const ushort_t* p = base + (size_t)(lane & 15) * stride + ((lane >> 4) << 3);
    Frag f;
    f.q[0] = *(const uint4*)(p);
    f.q[1] = *(const uint4*)(p + 16);
    return f.v;
}

__device__ __forceinline__ v8f wmma_bf16(v16bf a, v16bf b, v8f c) {
    return __builtin_amdgcn_wmma_f32_16x16x32_bf16(false, a, false, b, (short)0, c, false, false);
}

// ---------------------------------------------------------------------------
// Stage 1: f32 -> bf16 conversion
// ---------------------------------------------------------------------------
__global__ void cvt_f32_to_bf16(const float* __restrict__ src,
                                ushort_t* __restrict__ dst, int n) {
    int i = blockIdx.x * blockDim.x + threadIdx.x;
    if (i < n) dst[i] = f2bf(src[i]);
}

// ---------------------------------------------------------------------------
// Stage 2/4: NT GEMM  C[M,N] = (A[M,K](bf16) * B[N,K](bf16)^T + bias) * oscale
//   MODE 0: write f32 to Cout[m*N + n]                  (final projection)
//   MODE 1: write bf16 to [B,H,S,HD] layout             (Q, K projections)
//   MODE 2: write bf16 to [B,H,HD,S] layout (V^T)       (V projection)
// Block tile 128x128, 8 waves of 64x32 each, BK = 32, double-buffered LDS
// filled by async global->LDS copies.
// ---------------------------------------------------------------------------
#define BM 128
#define BN 128
#define BK 32
#define LDP 40   // padded LDS row stride (multiple of 8 -> 16B aligned)

template <int MODE>
__global__ __launch_bounds__(256)
void gemm_bf16(const ushort_t* __restrict__ A, const ushort_t* __restrict__ Bmat,
               const float* __restrict__ bias, void* __restrict__ Cout,
               int M, int N, int K, float oscale) {
    __shared__ ushort_t Als[2][BM][LDP];
    __shared__ ushort_t Bls[2][BN][LDP];

    const int tid  = threadIdx.x;
    const int lane = tid & 31;
    const int wid  = tid >> 5;
    const int wm   = wid >> 2;                 // 0..1  (64 rows each)
    const int wn   = wid & 3;                  // 0..3  (32 cols each)
    const int m0   = blockIdx.y * BM;
    const int n0   = blockIdx.x * BN;

    v8f vzero = {};
    Acc acc[4][2];
#pragma unroll
    for (int i = 0; i < 4; i++)
#pragma unroll
        for (int j = 0; j < 2; j++) acc[i][j].v = vzero;

    const int lrow = tid >> 1;                 // 0..127
    const int lseg = (tid & 1) << 4;           // 0 or 16 elements

    const ushort_t* gA = A    + (size_t)(m0 + lrow) * K + lseg;
    const ushort_t* gB = Bmat + (size_t)(n0 + lrow) * K + lseg;

    const int NIT = K / BK;

    // prologue: fill buffer 0
    cp16B(&Als[0][lrow][lseg],     gA);
    cp16B(&Als[0][lrow][lseg + 8], gA + 8);
    cp16B(&Bls[0][lrow][lseg],     gB);
    cp16B(&Bls[0][lrow][lseg + 8], gB + 8);

    for (int it = 0; it < NIT; ++it) {
        const int buf = it & 1;
        if (it + 1 < NIT) {
            const int kk = (it + 1) * BK;
            __builtin_prefetch(gA + kk + BK, 0, 0);
            cp16B(&Als[buf ^ 1][lrow][lseg],     gA + kk);
            cp16B(&Als[buf ^ 1][lrow][lseg + 8], gA + kk + 8);
            cp16B(&Bls[buf ^ 1][lrow][lseg],     gB + kk);
            cp16B(&Bls[buf ^ 1][lrow][lseg + 8], gB + kk + 8);
            WAIT_ASYNC(4);                     // current buffer's 4 copies done
        } else {
            WAIT_ASYNC(0);
        }
        __syncthreads();

        v16bf af[4], bfr[2];
#pragma unroll
        for (int i = 0; i < 4; i++)
            af[i] = load_fragA(&Als[buf][wm * 64 + i * 16][0], LDP, lane);
#pragma unroll
        for (int j = 0; j < 2; j++)
            bfr[j] = load_fragB(&Bls[buf][wn * 32 + j * 16][0], LDP, lane);
#pragma unroll
        for (int i = 0; i < 4; i++)
#pragma unroll
            for (int j = 0; j < 2; j++)
                acc[i][j].v = wmma_bf16(af[i], bfr[j], acc[i][j].v);
        __syncthreads();                       // done reading buf before reuse
    }

    // C/D layout: VGPR r, lanes 0-15 -> row r, col lane; lanes 16-31 -> row 8+r.
    const int rbase = 8 * (lane >> 4);
    const int cl    = lane & 15;
#pragma unroll
    for (int i = 0; i < 4; i++) {
#pragma unroll
        for (int j = 0; j < 2; j++) {
            const int tm = m0 + wm * 64 + i * 16;
            const int tn = n0 + wn * 32 + j * 16;
#pragma unroll
            for (int r = 0; r < 8; r++) {
                const int row = tm + rbase + r;
                const int col = tn + cl;
                const float val = (acc[i][j].f[r] + bias[col]) * oscale;
                if (MODE == 0) {
                    ((float*)Cout)[(size_t)row * N + col] = val;
                } else if (MODE == 1) {
                    const int b = row / S_, s = row % S_;
                    const int h = col >> 6, hd = col & 63;
                    ((ushort_t*)Cout)[((((size_t)b * H_ + h) * S_ + s) << 6) + hd] = f2bf(val);
                } else {
                    const int b = row / S_, s = row % S_;
                    const int h = col >> 6, hd = col & 63;
                    ((ushort_t*)Cout)[(((size_t)b * H_ + h) * HD_ + hd) * S_ + s] = f2bf(val);
                }
            }
        }
    }
}

// ---------------------------------------------------------------------------
// Stage 3: flash attention.  Q:[BH,S,64] bf16 (pre-scaled by 1/8),
// K:[BH,S,64] bf16, Vt:[BH,64,S] bf16 -> O:[B,S,D] bf16.
// One wave per 16 query rows, streams 32 KV/iter.  Row-sums come from a
// WMMA against an all-ones B fragment (accumulated & rescaled like O),
// so only the row-max needs a cross-lane reduction.
// ---------------------------------------------------------------------------
__global__ __launch_bounds__(128)
void flash_attn(const ushort_t* __restrict__ Q, const ushort_t* __restrict__ Kh,
                const ushort_t* __restrict__ Vt, ushort_t* __restrict__ O) {
    __shared__ ushort_t P[4][16][32];          // per-wave probability tile (bf16)

    const int tid  = threadIdx.x;
    const int lane = tid & 31;
    const int wid  = tid >> 5;
    const int b    = blockIdx.z;
    const int h    = blockIdx.y;
    const size_t bh = (size_t)b * H_ + h;
    const int q0   = (blockIdx.x * 4 + wid) * 16;

    // Q tile: 16 rows x 64 K -> two 16x32 A-fragments (held for whole loop)
    const ushort_t* qp = Q + (bh * S_ + q0) * HD_;
    const v16bf aq0 = load_fragA(qp,      HD_, lane);
    const v16bf aq1 = load_fragA(qp + 32, HD_, lane);

    // all-ones bf16 B fragment for row-sum accumulation
    Frag onesf;
    onesf.q[0] = make_uint4(0x3F803F80u, 0x3F803F80u, 0x3F803F80u, 0x3F803F80u);
    onesf.q[1] = onesf.q[0];
    const v16bf bones = onesf.v;

    v8f vzero = {};
    Acc o[4], osum;
#pragma unroll
    for (int t = 0; t < 4; t++) o[t].v = vzero;
    osum.v = vzero;
    float mrow[8];
#pragma unroll
    for (int r = 0; r < 8; r++) mrow[r] = -3.0e38f;

    const int prow = 8 * (lane >> 4);
    const int pcol = lane & 15;

    const ushort_t* kbase = Kh + bh * (size_t)S_ * HD_;
    const ushort_t* vbase = Vt + bh * (size_t)HD_ * S_;

    for (int n0 = 0; n0 < S_; n0 += 32) {
        // ---- scores: S[16x32] = Qs(16x64) * K(n0..n0+31, 0..63)^T ----
        Acc s0, s1;
        {
            const ushort_t* kp0 = kbase + (size_t)n0 * HD_;
            const ushort_t* kp1 = kp0 + 16 * HD_;
            const v16bf b00 = load_fragB(kp0,      HD_, lane);  // cols n0..+15, K 0..31
            const v16bf b01 = load_fragB(kp0 + 32, HD_, lane);  // cols n0..+15, K 32..63
            const v16bf b10 = load_fragB(kp1,      HD_, lane);
            const v16bf b11 = load_fragB(kp1 + 32, HD_, lane);
            s0.v = wmma_bf16(aq0, b00, vzero);
            s0.v = wmma_bf16(aq1, b01, s0.v);
            s1.v = wmma_bf16(aq0, b10, vzero);
            s1.v = wmma_bf16(aq1, b11, s1.v);
        }

        // ---- online softmax: row max via half-wave xor tree, no sum tree ----
#pragma unroll
        for (int r = 0; r < 8; r++) {
            const float x0 = s0.f[r];
            const float x1 = s1.f[r];
            float mx = fmaxf(x0, x1);
#pragma unroll
            for (int d = 8; d >= 1; d >>= 1) mx = fmaxf(mx, __shfl_xor(mx, d, 32));
            const float mn    = fmaxf(mrow[r], mx);
            const float alpha = __expf(mrow[r] - mn);
            const float p0    = __expf(x0 - mn);
            const float p1    = __expf(x1 - mn);
            mrow[r] = mn;
            osum.f[r] *= alpha;
#pragma unroll
            for (int t = 0; t < 4; t++) o[t].f[r] *= alpha;
            // C-layout -> LDS so P can be reloaded in A-fragment layout
            P[wid][prow + r][pcol]      = f2bf(p0);
            P[wid][prow + r][pcol + 16] = f2bf(p1);
        }

        // ---- O += P(16x32) * V; l += P(16x32) * ones ----
        const v16bf ap = load_fragA(&P[wid][0][0], 32, lane);
        osum.v = wmma_bf16(ap, bones, osum.v);
#pragma unroll
        for (int t = 0; t < 4; t++) {
            const v16bf bv = load_fragB(vbase + (size_t)(t * 16) * S_ + n0,
                                        (size_t)S_, lane);
            o[t].v = wmma_bf16(ap, bv, o[t].v);
        }
    }

    // ---- epilogue: normalize, store bf16 to [B,S,D] (D index = h*64+hd) ----
    // osum's element for this lane is exactly the row sum (identical across
    // the 16 lanes of each half), so no reduction is needed.
#pragma unroll
    for (int r = 0; r < 8; r++) {
        const float inv = 1.0f / osum.f[r];
        const int row = q0 + prow + r;
        const size_t obase = ((size_t)b * S_ + row) * D_ + (size_t)h * HD_;
#pragma unroll
        for (int t = 0; t < 4; t++)
            O[obase + t * 16 + pcol] = f2bf(o[t].f[r] * inv);
    }
}

// ---------------------------------------------------------------------------
// Host orchestration
// ---------------------------------------------------------------------------
extern "C" void kernel_launch(void* const* d_in, const int* in_sizes, int n_in,
                              void* d_out, int out_size, void* d_ws, size_t ws_size,
                              hipStream_t stream) {
    const float* q  = (const float*)d_in[0];
    const float* k  = (const float*)d_in[1];
    const float* v  = (const float*)d_in[2];
    const float* Wq = (const float*)d_in[3];
    const float* bq = (const float*)d_in[4];
    const float* Wk = (const float*)d_in[5];
    const float* bk = (const float*)d_in[6];
    const float* Wv = (const float*)d_in[7];
    const float* bv = (const float*)d_in[8];
    const float* Wo = (const float*)d_in[9];
    const float* bo = (const float*)d_in[10];
    float* out = (float*)d_out;

    const size_t XS  = (size_t)B_ * S_ * D_;   // 8,388,608 elements
    const size_t WSZ = (size_t)D_ * D_;        // 1,048,576 elements

    ushort_t* w   = (ushort_t*)d_ws;
    ushort_t* xq  = w;
    ushort_t* xk  = xq  + XS;
    ushort_t* xv  = xk  + XS;
    ushort_t* wqb = xv  + XS;
    ushort_t* wkb = wqb + WSZ;
    ushort_t* wvb = wkb + WSZ;
    ushort_t* wob = wvb + WSZ;
    ushort_t* qh  = wob + WSZ;                 // [B,H,S,HD], pre-scaled by 1/8
    ushort_t* kh  = qh  + XS;                  // [B,H,S,HD]
    ushort_t* vt  = kh  + XS;                  // [B,H,HD,S]
    ushort_t* oh  = vt  + XS;                  // [B,S,D]

    const dim3 cb(256);
    cvt_f32_to_bf16<<<dim3((unsigned)(XS / 256)), cb, 0, stream>>>(q, xq, (int)XS);
    cvt_f32_to_bf16<<<dim3((unsigned)(XS / 256)), cb, 0, stream>>>(k, xk, (int)XS);
    cvt_f32_to_bf16<<<dim3((unsigned)(XS / 256)), cb, 0, stream>>>(v, xv, (int)XS);
    cvt_f32_to_bf16<<<dim3((unsigned)(WSZ / 256)), cb, 0, stream>>>(Wq, wqb, (int)WSZ);
    cvt_f32_to_bf16<<<dim3((unsigned)(WSZ / 256)), cb, 0, stream>>>(Wk, wkb, (int)WSZ);
    cvt_f32_to_bf16<<<dim3((unsigned)(WSZ / 256)), cb, 0, stream>>>(Wv, wvb, (int)WSZ);
    cvt_f32_to_bf16<<<dim3((unsigned)(WSZ / 256)), cb, 0, stream>>>(Wo, wob, (int)WSZ);

    const dim3 gg(D_ / BN, (B_ * S_) / BM), gb(256);
    // fold softmax scale 1/sqrt(64) into the Q projection (incl. bias)
    gemm_bf16<1><<<gg, gb, 0, stream>>>(xq, wqb, bq, qh, B_ * S_, D_, D_, 0.125f);
    gemm_bf16<1><<<gg, gb, 0, stream>>>(xk, wkb, bk, kh, B_ * S_, D_, D_, 1.0f);
    gemm_bf16<2><<<gg, gb, 0, stream>>>(xv, wvb, bv, vt, B_ * S_, D_, D_, 1.0f);

    flash_attn<<<dim3(S_ / 64, H_, B_), dim3(128), 0, stream>>>(qh, kh, vt, oh);

    gemm_bf16<0><<<gg, gb, 0, stream>>>(oh, wob, bo, out, B_ * S_, D_, D_, 1.0f);
}
